// HopeAttention_56513179680992
// MI455X (gfx1250) — compile-verified, matrix operand order checked by reading, not run on hardware
//
#include <hip/hip_runtime.h>

// ---------------------------------------------------------------------------
// HopeAttention for MI455X (gfx1250): f16 WMMA compute, b128 fragment loads,
// pre-transposed weights, prefetch, no predicated staging.
// B=4, S=2048, D_MODEL=1024, H=16, KV=4, HD=64, KV_DIM=256, CHUNK=64
// ---------------------------------------------------------------------------

typedef _Float16 v16h __attribute__((ext_vector_type(16)));
typedef float    v8f  __attribute__((ext_vector_type(8)));
typedef float    fx8  __attribute__((ext_vector_type(8)));
typedef _Float16 hx8  __attribute__((ext_vector_type(8)));

#define WMMA16(a, b, c) \
  __builtin_amdgcn_wmma_f32_16x16x32_f16(false, (a), false, (b), (short)0, (c), false, false)

__device__ __forceinline__ v8f v8f_zero() {
  v8f z;
#pragma unroll
  for (int i = 0; i < 8; ++i) z[i] = 0.f;
  return z;
}
__device__ __forceinline__ fx8 fx8_zero() {
  fx8 z;
#pragma unroll
  for (int i = 0; i < 8; ++i) z[i] = 0.f;
  return z;
}
__device__ __forceinline__ fx8 splat8(float x) {
  fx8 v;
#pragma unroll
  for (int i = 0; i < 8; ++i) v[i] = x;
  return v;
}
__device__ __forceinline__ float hsum8(fx8 v) {
  float s = 0.f;
#pragma unroll
  for (int i = 0; i < 8; ++i) s += v[i];
  return s;
}
__device__ __forceinline__ hx8 cvt8(fx8 f) {
  return __builtin_convertvector(f, hx8);
}
// Build a 16-half fragment from two 16B-aligned 8-half chunks.
__device__ __forceinline__ v16h frag16(const _Float16* lo, const _Float16* hi) {
  hx8 a = *(const hx8*)lo;
  hx8 b = *(const hx8*)hi;
  return __builtin_shufflevector(a, b, 0, 1, 2, 3, 4, 5, 6, 7, 8, 9, 10, 11, 12, 13, 14, 15);
}
__device__ __forceinline__ float sigmoidf_(float x) {
  return 1.f / (1.f + __expf(-x));
}

// ---------------------------------------------------------------------------
// LDS-tiled fp32 transpose: in [K][N] -> out [N][K]. Grid (K/32, N/32), 256 thr.
// ---------------------------------------------------------------------------
__global__ __launch_bounds__(256) void transpose_f32(
    const float* __restrict__ in, float* __restrict__ out, int K, int N) {
  __shared__ float tile[32][33];
  const int kb = blockIdx.x * 32, nb = blockIdx.y * 32;
  const int t = threadIdx.x & 255;
  const int tx = t & 31, ty = t >> 5;  // 8 rows per pass
#pragma unroll
  for (int r = ty; r < 32; r += 8) tile[r][tx] = in[(size_t)(kb + r) * N + nb + tx];
  __syncthreads();
#pragma unroll
  for (int r = ty; r < 32; r += 8) out[(size_t)(nb + r) * K + kb + tx] = tile[tx][r];
}

// ---------------------------------------------------------------------------
// GEMM: C[M,N] = A[M,K] @ B[K,N] with B supplied TRANSPOSED (BT: [N][K]).
// 128 threads (4 waves), 64x64 tile, K-step 32. A and B staging are the same
// fully vectorized path: global b128x2 -> cvt_pk -> ds b128. Fragments are
// 2x ds_load_b128. Prefetch of the next k-slab overlaps the WMMA phase.
// ---------------------------------------------------------------------------
__global__ __launch_bounds__(128) void gemm64_wmma(
    const float* __restrict__ A, const float* __restrict__ BT,
    float* __restrict__ C, int K, int N) {
  __shared__ __attribute__((aligned(16))) _Float16 As[64][40];   // [m][k]
  __shared__ __attribute__((aligned(16))) _Float16 BsT[64][40];  // [n][k]
  const int tid = threadIdx.x & 127;  // provable range: kills staging predication
  const int wv = tid >> 5, lane = tid & 31;
  const int hl = lane >> 4, lm = lane & 15;
  const long m0 = (long)blockIdx.x * 64;
  const int n0 = blockIdx.y * 64;

  v8f acc[4];
#pragma unroll
  for (int t = 0; t < 4; ++t) acc[t] = v8f_zero();

  for (int k0 = 0; k0 < K; k0 += 32) {
#pragma unroll
    for (int it = 0; it < 2; ++it) {
      const int e = tid + it * 128;        // 0..255
      const int r = e >> 2, c = (e & 3) * 8;
      const float* ap = A + (m0 + r) * K + k0 + c;
      const float* bp = BT + (size_t)(n0 + r) * K + k0 + c;
      *(hx8*)&As[r][c] = cvt8(*(const fx8*)ap);
      *(hx8*)&BsT[r][c] = cvt8(*(const fx8*)bp);
      if (k0 + 32 < K) {
        __builtin_prefetch((const void*)(ap + 32), 0, 3);
        __builtin_prefetch((const void*)(bp + 32), 0, 3);
      }
    }
    __syncthreads();

    const int ar = wv * 16 + lm;
    v16h a = frag16(&As[ar][hl * 8], &As[ar][16 + hl * 8]);
#pragma unroll
    for (int t = 0; t < 4; ++t) {
      v16h b = frag16(&BsT[t * 16 + lm][hl * 8], &BsT[t * 16 + lm][16 + hl * 8]);
      acc[t] = WMMA16(a, b, acc[t]);
    }
    __syncthreads();
  }

#pragma unroll
  for (int t = 0; t < 4; ++t) {
#pragma unroll
    for (int r = 0; r < 8; ++r) {
      long row = m0 + wv * 16 + hl * 8 + r;
      C[row * N + n0 + t * 16 + lm] = acc[t][r];
    }
  }
}

// ---------------------------------------------------------------------------
// Chunked delta-rule memory scan. 8 blocks (2 inst x 4 batch), 256 threads.
// Retrieval GEMM via WMMA with LDS-staged f16 M k-slab (reused by 4 waves);
// vectorized fp32 epilogue for the delta update / gate / norm clamp.
// ---------------------------------------------------------------------------
__global__ __launch_bounds__(256) void mem_scan(
    const float* __restrict__ x, const float* __restrict__ M_init,
    const float* __restrict__ eta_w, const float* __restrict__ eta_b,
    const float* __restrict__ alpha_w, const float* __restrict__ alpha_b,
    const float* __restrict__ gate_w, const float* __restrict__ gate_b,
    float* __restrict__ Mst_all, float* __restrict__ Mtmp_all,
    float* __restrict__ mem_out) {
  const int inst = blockIdx.x >> 2;
  const int bat = blockIdx.x & 3;
  const int tid = threadIdx.x & 255;  // provable range
  const int wv = tid >> 5, lane = tid & 31;
  const int hl = lane >> 4, lm = lane & 15;
  const int rw = (wv & 3) * 16;    // row stripe within 64-token chunk
  const int cb = (wv >> 2) * 128;  // column base within 256

  float* Mi = Mst_all + (size_t)blockIdx.x * (256 * 1024);
  float* Mt = Mtmp_all + (size_t)blockIdx.x * (256 * 1024);
  const float* Minit = M_init + (size_t)inst * (256 * 1024);
  const float* ew = eta_w + inst * 1024;
  const float* aw = alpha_w + inst * 1024;
  const float* gw = gate_w + inst * 1024;
  const float eb = eta_b[inst], ab = alpha_b[inst], gb = gate_b[inst];
  float* outp = mem_out + (size_t)blockIdx.x * (2048 * 256);
  const float* xb = x + (size_t)bat * (2048 * 1024);

  __shared__ __attribute__((aligned(16))) _Float16 As[64][40];   // chunk [m][k]
  __shared__ __attribute__((aligned(16))) _Float16 Ms[256][40];  // M [n][k] slab
  __shared__ float vt[256];
  __shared__ __attribute__((aligned(32))) float kmean[1024];
  __shared__ float err[256];
  __shared__ float p0[256], p1[256], p2[256];
  __shared__ float norms[64], rnorm[64], etaTok[64], alTok[64];
  __shared__ float scal[4];  // eta, alpha, gate, normScale

  // init M state from M_init (vectorized copy)
  for (int e = tid * 8; e < 256 * 1024; e += 256 * 8)
    *(fx8*)&Mi[e] = *(const fx8*)&Minit[e];
  __syncthreads();

  for (int ch = 0; ch < 32; ++ch) {
    const float* xc = xb + (size_t)ch * 64 * 1024;
    vt[tid] = 0.f;
    __syncthreads();

    // ---- retrieval GEMM: out[64,256] = xc[64,1024] @ Mi^T ----
    v8f acc[8];
#pragma unroll
    for (int t = 0; t < 8; ++t) acc[t] = v8f_zero();

    for (int k0 = 0; k0 < 1024; k0 += 32) {
      // stage chunk A slab: 64 rows x 4 chunks, 1 chunk per thread
      {
        const int r = tid >> 2, c = (tid & 3) * 8;
        *(hx8*)&As[r][c] = cvt8(*(const fx8*)(xc + (size_t)r * 1024 + k0 + c));
      }
      // stage M slab: one 32-wide row per thread, fully vectorized
      {
        const float* Mr = Mi + (size_t)tid * 1024 + k0;
#pragma unroll
        for (int c = 0; c < 4; ++c)
          *(hx8*)&Ms[tid][c * 8] = cvt8(*(const fx8*)(Mr + c * 8));
      }
      __syncthreads();

      v16h a = frag16(&As[rw + lm][hl * 8], &As[rw + lm][16 + hl * 8]);
#pragma unroll
      for (int t = 0; t < 8; ++t) {
        const int n = cb + t * 16 + lm;
        v16h b = frag16(&Ms[n][hl * 8], &Ms[n][16 + hl * 8]);
        acc[t] = WMMA16(a, b, acc[t]);
      }
      __syncthreads();
    }

    // write out + accumulate per-column sums (v_target * 64)
#pragma unroll
    for (int t = 0; t < 8; ++t) {
      float s = 0.f;
#pragma unroll
      for (int r = 0; r < 8; ++r) {
        int row = ch * 64 + rw + hl * 8 + r;
        outp[(size_t)row * 256 + cb + t * 16 + lm] = acc[t][r];
        s += acc[t][r];
      }
      s += __shfl_xor(s, 16, 32);
      if (hl == 0) atomicAdd(&vt[cb + t * 16 + lm], s);
    }
    __syncthreads();

    // ---- per-token norm^2 and eta/alpha dot partials (vectorized) ----
    {
      const int tok = tid >> 2, seg = tid & 3;
      const float* xr = xc + (size_t)tok * 1024 + seg * 256;
      const float* ewp = ew + seg * 256;
      const float* awp = aw + seg * 256;
      fx8 s2 = fx8_zero(), se = fx8_zero(), sa = fx8_zero();
      for (int d = 0; d < 256; d += 8) {
        fx8 v = *(const fx8*)(xr + d);
        s2 += v * v;
        se += v * *(const fx8*)(ewp + d);
        sa += v * *(const fx8*)(awp + d);
      }
      p0[tid] = hsum8(s2);
      p1[tid] = hsum8(se);
      p2[tid] = hsum8(sa);
    }
    __syncthreads();
    if (tid < 64) {
      float s2 = p0[4 * tid] + p0[4 * tid + 1] + p0[4 * tid + 2] + p0[4 * tid + 3];
      float se = p1[4 * tid] + p1[4 * tid + 1] + p1[4 * tid + 2] + p1[4 * tid + 3];
      float sa = p2[4 * tid] + p2[4 * tid + 1] + p2[4 * tid + 2] + p2[4 * tid + 3];
      float nn = sqrtf(s2);
      norms[tid] = nn;
      rnorm[tid] = 1.f / fmaxf(nn, 1e-5f);
      etaTok[tid] = sigmoidf_(se + eb);
      alTok[tid] = sigmoidf_(sa + ab);
    }
    __syncthreads();
    if (tid == 0) {
      float me = 0.f, ma = 0.f;
      for (int i = 0; i < 64; ++i) { me += etaTok[i]; ma += alTok[i]; }
      scal[0] = (me / 64.f) * 0.2f;         // * MAX_LR
      scal[1] = 0.5f + (ma / 64.f) * 0.5f;  // MIN_DECAY + mean*(1-MIN_DECAY)
    }
    __syncthreads();

    // ---- k_mean (L2-normalized keys, mean over tokens) ----
    for (int d = tid; d < 1024; d += 256) {
      float s = 0.f;
      for (int t2 = 0; t2 < 64; ++t2) s += xc[(size_t)t2 * 1024 + d] * rnorm[t2];
      kmean[d] = s * (1.f / 64.f);
    }
    __syncthreads();

    // ---- memory gate ----
    {
      float g = 0.f;
      for (int d = tid; d < 1024; d += 256) g += kmean[d] * gw[d];
      p0[tid] = g;
    }
    __syncthreads();
    if (tid == 0) {
      float g = 0.f;
      for (int i = 0; i < 256; ++i) g += p0[i];
      scal[2] = sigmoidf_(g + gb);
    }
    __syncthreads();

    // ---- Mk and err (vectorized dot) ----
    {
      const float* Mr = Mi + (size_t)tid * 1024;
      fx8 a8 = fx8_zero();
      for (int d = 0; d < 1024; d += 8)
        a8 += *(const fx8*)(Mr + d) * *(const fx8*)&kmean[d];
      err[tid] = vt[tid] / 64.f - hsum8(a8);
    }
    __syncthreads();

    // ---- delta update + gate blend + Frobenius norm^2 (vectorized) ----
    {
      float eta = scal[0], alpha = scal[1], gate = scal[2];
      fx8 nrm = fx8_zero();
      for (int e = tid * 8; e < 256 * 1024; e += 2048) {
        int o = e >> 10, d = e & 1023;
        fx8 mold = *(const fx8*)&Mi[e];
        fx8 mnew = splat8(alpha) * mold + splat8(eta * err[o]) * *(const fx8*)&kmean[d];
        mnew = splat8(gate) * mnew + splat8(1.f - gate) * mold;
        *(fx8*)&Mt[e] = mnew;
        nrm += mnew * mnew;
      }
      p0[tid] = hsum8(nrm);
    }
    __syncthreads();
    if (tid == 0) {
      float s = 0.f;
      for (int i = 0; i < 256; ++i) s += p0[i];
      scal[3] = fminf(30.f / (sqrtf(s) + 1e-6f), 1.f);  // MAX_NORM clamp
    }
    __syncthreads();
    {
      fx8 s = splat8(scal[3]);
      for (int e = tid * 8; e < 256 * 1024; e += 2048)
        *(fx8*)&Mi[e] = *(const fx8*)&Mt[e] * s;
    }
    __syncthreads();
  }
}

// ---------------------------------------------------------------------------
// Fuse: k = k_lin + k_mem, v = v_lin + v_mem, RoPE(q,k), pack f16.
// q -> [B,16,S,64], k -> [B,4,S,64], v -> TRANSPOSED [B,4,64,S] so that the
// P@V B-fragments in flash attention are contiguous 16B global loads.
// ---------------------------------------------------------------------------
__global__ __launch_bounds__(128) void rope_pack(
    const float* __restrict__ q_lin, const float* __restrict__ k_lin,
    const float* __restrict__ v_lin, const float* __restrict__ mem_out,
    _Float16* __restrict__ qh, _Float16* __restrict__ kh, _Float16* __restrict__ vhT) {
  const int bs = blockIdx.x;
  const int b = bs >> 11;
  const int s = bs & 2047;
  const int tid = threadIdx.x & 127;  // provable range
  const float* kmem = mem_out + ((size_t)(0 * 4 + b) * 2048 + s) * 256;
  const float* vmem = mem_out + ((size_t)(1 * 4 + b) * 2048 + s) * 256;
  const float* qr = q_lin + ((size_t)b * 2048 + s) * 1024;
  const float* kr = k_lin + ((size_t)b * 2048 + s) * 256;
  const float* vr = v_lin + ((size_t)b * 2048 + s) * 256;

  // q: 16 heads x 32 rotation pairs (4 per thread)
#pragma unroll
  for (int it = 0; it < 4; ++it) {
    const int idx = tid + it * 128;
    const int h = idx >> 5, i = idx & 31;
    float inv = __powf(500000.0f, -(float)(2 * i) / 64.0f);
    float sn, cs;
    __sincosf((float)s * inv, &sn, &cs);
    float x1 = qr[h * 64 + i];
    float x2 = qr[h * 64 + i + 32];
    size_t base = ((size_t)(b * 16 + h) * 2048 + s) * 64;
    qh[base + i] = (_Float16)(x1 * cs - x2 * sn);
    qh[base + i + 32] = (_Float16)(x2 * cs + x1 * sn);
  }
  // k: 4 heads x 32 pairs (add memory, then RoPE)
  {
    const int h = tid >> 5, i = tid & 31;
    float inv = __powf(500000.0f, -(float)(2 * i) / 64.0f);
    float sn, cs;
    __sincosf((float)s * inv, &sn, &cs);
    float x1 = kr[h * 64 + i] + kmem[h * 64 + i];
    float x2 = kr[h * 64 + i + 32] + kmem[h * 64 + i + 32];
    size_t base = ((size_t)(b * 4 + h) * 2048 + s) * 64;
    kh[base + i] = (_Float16)(x1 * cs - x2 * sn);
    kh[base + i + 32] = (_Float16)(x2 * cs + x1 * sn);
  }
  // v: add memory, no RoPE; write d-major [B,4,64,S] (2 per thread)
#pragma unroll
  for (int it = 0; it < 2; ++it) {
    const int idx = tid + it * 128;
    const int h = idx >> 6, d = idx & 63;
    float v = vr[h * 64 + d] + vmem[h * 64 + d];
    vhT[((size_t)(b * 4 + h) * 64 + d) * 2048 + s] = (_Float16)v;
  }
}

// ---------------------------------------------------------------------------
// Flash attention, causal + GQA (rep=4). Grid (S/64, 16, 4), 128 threads.
// All fragments are 2x16B loads; P reshaped through per-wave LDS tile.
// ---------------------------------------------------------------------------
__global__ __launch_bounds__(128) void flash_attn(
    const _Float16* __restrict__ qh, const _Float16* __restrict__ kh,
    const _Float16* __restrict__ vhT, float* __restrict__ attn_out) {
  const int qb = blockIdx.x;
  const int h = blockIdx.y;
  const int b = blockIdx.z;
  const int kvh = h >> 2;
  const int tid = threadIdx.x & 127;
  const int wv = tid >> 5, lane = tid & 31;
  const int hl = lane >> 4, lm = lane & 15;

  const _Float16* Q = qh + ((size_t)(b * 16 + h) * 2048 + qb * 64) * 64;
  const _Float16* Kp = kh + (size_t)(b * 4 + kvh) * 2048 * 64;
  const _Float16* Vt = vhT + (size_t)(b * 4 + kvh) * 64 * 2048;

  __shared__ __attribute__((aligned(16))) _Float16 P[4][16][72];

  // Q fragments: this wave's 16 rows, two k-slabs (d 0..31, 32..63)
  const _Float16* Qr = Q + (size_t)(wv * 16 + lm) * 64;
  v16h qa0 = frag16(Qr + hl * 8, Qr + 16 + hl * 8);
  v16h qa1 = frag16(Qr + 32 + hl * 8, Qr + 48 + hl * 8);

  v8f o[4];
#pragma unroll
  for (int t = 0; t < 4; ++t) o[t] = v8f_zero();
  float mrow[8], lrow[8];
#pragma unroll
  for (int r = 0; r < 8; ++r) { mrow[r] = -1e30f; lrow[r] = 0.f; }

  for (int j = 0; j <= qb; ++j) {
    // ---- scores: S = Q @ K^T ----
    v8f sc[4];
#pragma unroll
    for (int t = 0; t < 4; ++t) sc[t] = v8f_zero();
#pragma unroll
    for (int t = 0; t < 4; ++t) {
      const _Float16* Kr = Kp + (size_t)(j * 64 + t * 16 + lm) * 64;
      v16h kb0 = frag16(Kr + hl * 8, Kr + 16 + hl * 8);
      v16h kb1 = frag16(Kr + 32 + hl * 8, Kr + 48 + hl * 8);
      sc[t] = WMMA16(qa0, kb0, sc[t]);
      sc[t] = WMMA16(qa1, kb1, sc[t]);
    }

    // ---- scale, causal mask, online softmax ----
#pragma unroll
    for (int r = 0; r < 8; ++r) {
      int qrow = qb * 64 + wv * 16 + hl * 8 + r;
      float mx = -1e30f;
#pragma unroll
      for (int t = 0; t < 4; ++t) {
        float v = sc[t][r] * 0.125f;  // 1/sqrt(64)
        int kcol = j * 64 + t * 16 + lm;
        if (kcol > qrow) v = -1e30f;
        sc[t][r] = v;
        mx = fmaxf(mx, v);
      }
#pragma unroll
      for (int d = 1; d < 16; d <<= 1) mx = fmaxf(mx, __shfl_xor(mx, d, 32));
      float mnew = fmaxf(mrow[r], mx);
      float corr = __expf(mrow[r] - mnew);
      mrow[r] = mnew;
      float rs = 0.f;
#pragma unroll
      for (int t = 0; t < 4; ++t) {
        float p = __expf(sc[t][r] - mnew);
        sc[t][r] = p;
        rs += p;
      }
#pragma unroll
      for (int d = 1; d < 16; d <<= 1) rs += __shfl_xor(rs, d, 32);
      lrow[r] = lrow[r] * corr + rs;
#pragma unroll
      for (int t = 0; t < 4; ++t) o[t][r] *= corr;
    }

    // ---- P through per-wave LDS tile: D-frag -> A-frag reshape ----
#pragma unroll
    for (int t = 0; t < 4; ++t)
#pragma unroll
      for (int r = 0; r < 8; ++r)
        P[wv][hl * 8 + r][t * 16 + lm] = (_Float16)sc[t][r];

    const _Float16* Pr = &P[wv][lm][0];
    v16h pa0 = frag16(Pr + hl * 8, Pr + 16 + hl * 8);
    v16h pa1 = frag16(Pr + 32 + hl * 8, Pr + 48 + hl * 8);

    // ---- O += P @ V (V is d-major: contiguous along k) ----
#pragma unroll
    for (int t = 0; t < 4; ++t) {
      const _Float16* Vr = Vt + (size_t)(t * 16 + lm) * 2048 + j * 64;
      v16h vb0 = frag16(Vr + hl * 8, Vr + 16 + hl * 8);
      v16h vb1 = frag16(Vr + 32 + hl * 8, Vr + 48 + hl * 8);
      o[t] = WMMA16(pa0, vb0, o[t]);
      o[t] = WMMA16(pa1, vb1, o[t]);
    }
  }

  // ---- epilogue: normalize and write [B,S,H*64] ----
#pragma unroll
  for (int t = 0; t < 4; ++t)
#pragma unroll
    for (int r = 0; r < 8; ++r) {
      int qrow = qb * 64 + wv * 16 + hl * 8 + r;
      attn_out[((size_t)(b * 2048 + qrow)) * 1024 + h * 64 + t * 16 + lm] =
          o[t][r] / lrow[r];
    }
}

// ---------------------------------------------------------------------------
extern "C" void kernel_launch(void* const* d_in, const int* in_sizes, int n_in,
                              void* d_out, int out_size, void* d_ws, size_t ws_size,
                              hipStream_t stream) {
  (void)in_sizes; (void)n_in; (void)out_size; (void)ws_size;
  const float* x  = (const float*)d_in[0];
  const float* Wq = (const float*)d_in[1];
  const float* Wk = (const float*)d_in[2];
  const float* Wv = (const float*)d_in[3];
  const float* Wo = (const float*)d_in[4];
  const float* Mi = (const float*)d_in[5];
  const float* ew = (const float*)d_in[6];
  const float* eb = (const float*)d_in[7];
  const float* aw = (const float*)d_in[8];
  const float* ab = (const float*)d_in[9];
  const float* gw = (const float*)d_in[10];
  const float* gb = (const float*)d_in[11];
  float* out = (float*)d_out;

  char* w = (char*)d_ws;
  float* q_lin = (float*)w;    w += (size_t)8192 * 1024 * 4;
  float* k_lin = (float*)w;    w += (size_t)8192 * 256 * 4;
  float* v_lin = (float*)w;    w += (size_t)8192 * 256 * 4;
  float* memo  = (float*)w;    w += (size_t)2 * 8192 * 256 * 4;
  float* Mst   = (float*)w;    w += (size_t)8 * 256 * 1024 * 4;
  float* Mtmp  = (float*)w;    w += (size_t)8 * 256 * 1024 * 4;
  float* WqT   = (float*)w;    w += (size_t)1024 * 1024 * 4;
  float* WkT   = (float*)w;    w += (size_t)256 * 1024 * 4;
  float* WvT   = (float*)w;    w += (size_t)256 * 1024 * 4;
  float* WoT   = (float*)w;    w += (size_t)1024 * 1024 * 4;
  _Float16* qh = (_Float16*)w; w += (size_t)8192 * 1024 * 2;
  _Float16* kh = (_Float16*)w; w += (size_t)4 * 2048 * 64 * 2;
  _Float16* vhT = (_Float16*)w; w += (size_t)4 * 64 * 2048 * 2;
  float* attn = q_lin;  // reuse q_lin after rope_pack consumed it

  // One-time weight transposes ([K][N] -> [N][K]) for gather-free GEMM staging.
  transpose_f32<<<dim3(32, 32), 256, 0, stream>>>(Wq, WqT, 1024, 1024);
  transpose_f32<<<dim3(32, 8), 256, 0, stream>>>(Wk, WkT, 1024, 256);
  transpose_f32<<<dim3(32, 8), 256, 0, stream>>>(Wv, WvT, 1024, 256);
  transpose_f32<<<dim3(32, 32), 256, 0, stream>>>(Wo, WoT, 1024, 1024);

  mem_scan<<<8, 256, 0, stream>>>(x, Mi, ew, eb, aw, ab, gw, gb, Mst, Mtmp, memo);
  gemm64_wmma<<<dim3(128, 16), 128, 0, stream>>>(x, WqT, q_lin, 1024, 1024);
  gemm64_wmma<<<dim3(128, 4), 128, 0, stream>>>(x, WkT, k_lin, 1024, 256);
  gemm64_wmma<<<dim3(128, 4), 128, 0, stream>>>(x, WvT, v_lin, 1024, 256);
  rope_pack<<<8192, 128, 0, stream>>>(q_lin, k_lin, v_lin, memo, qh, kh, vhT);
  flash_attn<<<dim3(32, 16, 4), 128, 0, stream>>>(qh, kh, vhT, attn);
  gemm64_wmma<<<dim3(128, 16), 128, 0, stream>>>(attn, WoT, out, 1024, 1024);
}
